// Block_90065464197735
// MI455X (gfx1250) — compile-verified, matrix operand order checked by reading, not run on hardware
//
#include <hip/hip_runtime.h>
#include <cmath>

typedef __attribute__((ext_vector_type(16))) __bf16 v16bf;
typedef __attribute__((ext_vector_type(8)))  float  v8f;

// Problem constants (match reference)
constexpr int Bb  = 4;
constexpr int Ss  = 2048;
constexpr int Hh  = 1024;
constexpr int NH  = 16;
constexpr int HD  = 64;      // head dim
constexpr int Ii  = 4096;
constexpr int Mrows = Bb * Ss;           // 8192 token rows
constexpr float LN_EPS = 1e-5f;

// ---------------------------------------------------------------------------
// helpers
// ---------------------------------------------------------------------------
__device__ __forceinline__ int kbase(int i, int hi) {
    // 16-bit A/B fragment K index for VGPR i, half-wave hi (CDNA5 ISA 7.12.2)
    return ((i < 4) ? (2 * i) : (2 * i + 8)) + 8 * hi;
}

__device__ __forceinline__ float gelu_tanh(float x) {
    const float c = 0.7978845608028654f; // sqrt(2/pi)
    return 0.5f * x * (1.0f + tanhf(c * (x + 0.044715f * x * x * x)));
}

__device__ __forceinline__ v8f wmma_bf16(v16bf a, v16bf b, v8f c) {
    return __builtin_amdgcn_wmma_f32_16x16x32_bf16(false, a, false, b,
                                                   (short)0, c, false, false);
}

// low 32 bits of a generic pointer to LDS == LDS byte offset (flat aperture rule)
__device__ __forceinline__ unsigned lds_addr32(const void* p) {
    return (unsigned)(uintptr_t)p;
}

// async global->LDS 16-byte copy (per-lane addresses), tracked by ASYNCcnt
__device__ __forceinline__ void async_copy_b128(unsigned lds_byte_addr,
                                                const void* gptr) {
    unsigned long long ga = (unsigned long long)(uintptr_t)gptr;
    asm volatile("global_load_async_to_lds_b128 %0, %1, off"
                 :: "v"(lds_byte_addr), "v"(ga) : "memory");
}

__device__ __forceinline__ void wait_async0() { asm volatile("s_wait_asynccnt 0x0" ::: "memory"); }
__device__ __forceinline__ void wait_async2() { asm volatile("s_wait_asynccnt 0x2" ::: "memory"); }
__device__ __forceinline__ void wait_async3() { asm volatile("s_wait_asynccnt 0x3" ::: "memory"); }

// DPP16 cross-lane reductions over each 16-lane half (pure VALU, no LDS)
template <int CTRL>
__device__ __forceinline__ float dpp_movf(float x) {
    return __int_as_float(__builtin_amdgcn_update_dpp(
        0, __float_as_int(x), CTRL, 0xF, 0xF, true));
}
__device__ __forceinline__ float red16_max(float v) {
    v = fmaxf(v, dpp_movf<0xB1>(v));    // quad_perm [1,0,3,2]  (xor 1)
    v = fmaxf(v, dpp_movf<0x4E>(v));    // quad_perm [2,3,0,1]  (xor 2)
    v = fmaxf(v, dpp_movf<0x141>(v));   // row_half_mirror
    v = fmaxf(v, dpp_movf<0x140>(v));   // row_mirror
    return v;
}
__device__ __forceinline__ float red16_sum(float v) {
    v += dpp_movf<0xB1>(v);
    v += dpp_movf<0x4E>(v);
    v += dpp_movf<0x141>(v);
    v += dpp_movf<0x140>(v);
    return v;
}

// pack two bf16 from a 32-bit LDS/global read into fragment slots
__device__ __forceinline__ void put_pair(v16bf& f, int i, unsigned u) {
    f[2 * i]     = ((const __bf16*)&u)[0];
    f[2 * i + 1] = ((const __bf16*)&u)[1];
}

// ---------------------------------------------------------------------------
// f32 -> bf16 cast + transpose (weights):  in[K][N] -> out[N][K]
// ---------------------------------------------------------------------------
__global__ void cast_transpose_kernel(const float* __restrict__ in,
                                      __bf16* __restrict__ out, int K, int N) {
    int idx = blockIdx.x * 256 + threadIdx.x;
    if (idx < K * N) {
        int k = idx / N, n = idx - k * N;
        out[(size_t)n * K + k] = (__bf16)in[idx];
    }
}

// ---------------------------------------------------------------------------
// LayerNorm over H=1024, one block per row, bf16 output
// ---------------------------------------------------------------------------
__global__ __launch_bounds__(256)
void ln_kernel(const float* __restrict__ x, const float* __restrict__ g,
               const float* __restrict__ b, __bf16* __restrict__ out) {
    const int row = blockIdx.x;
    const int tid = threadIdx.x;
    const float4 v4 = ((const float4*)(x + (size_t)row * Hh))[tid];
    const float v[4] = {v4.x, v4.y, v4.z, v4.w};

    float s1 = 0.f, s2 = 0.f;
#pragma unroll
    for (int t = 0; t < 4; ++t) { s1 += v[t]; s2 += v[t] * v[t]; }

    __shared__ float r1[256];
    __shared__ float r2[256];
    r1[tid] = s1; r2[tid] = s2;
    __syncthreads();
    for (int st = 128; st > 0; st >>= 1) {
        if (tid < st) { r1[tid] += r1[tid + st]; r2[tid] += r2[tid + st]; }
        __syncthreads();
    }
    const float mu  = r1[0] * (1.0f / Hh);
    const float var = r2[0] * (1.0f / Hh) - mu * mu;
    const float rstd = rsqrtf(var + LN_EPS);
    __bf16* orow = out + (size_t)row * Hh;
#pragma unroll
    for (int t = 0; t < 4; ++t) {
        int c = tid * 4 + t;
        orow[c] = (__bf16)((v[t] - mu) * rstd * g[c] + b[c]);
    }
}

// ---------------------------------------------------------------------------
// Tiled WMMA GEMM with pre-transposed weights Wt[N][K].
// out[M,N] = epi(A_bf16[M,K] @ W[K,N] + bias [+resid])
// Block tile 128(M) x 64(N), BK=32, 256 threads = 8 waves, wave = 16 rows.
// Both tiles staged via async B128 copies, double-buffered LDS.
// MODE: 0 = f32 out, 1 = bf16 out, 2 = QKV split (qh/kh [b,h,s,d], vtr [b,h,d,s])
// ---------------------------------------------------------------------------
constexpr int BM = 128, BN = 64, BK = 32;
constexpr int AS_S = 40;   // A LDS row stride elems: 80 B (16B multiple)
constexpr int BT_S = 40;   // Bt LDS row stride elems: 80 B (16B multiple)

template <int MODE, bool GELU, bool RESID>
__global__ __launch_bounds__(256)
void gemm_wmma(const __bf16* __restrict__ A, const __bf16* __restrict__ Wt,
               const float* __restrict__ bias, const float* __restrict__ resid,
               void* __restrict__ out0, void* __restrict__ out1,
               void* __restrict__ out2, int Mdim, int Ndim, int Kdim) {
    __shared__ __align__(16) __bf16 As[2][BM * AS_S];   // row-major [m][k]
    __shared__ __align__(16) __bf16 Bt[2][BN * BT_S];   // n-major  [n][k]

    const int tid  = threadIdx.x;
    const int wave = tid >> 5;
    const int lane = tid & 31;
    const int m16  = lane & 15;
    const int hi   = lane >> 4;
    const int m0   = blockIdx.y * BM;
    const int n0   = blockIdx.x * BN;
    const unsigned as_base[2] = {lds_addr32(&As[0][0]), lds_addr32(&As[1][0])};
    const unsigned bt_base[2] = {lds_addr32(&Bt[0][0]), lds_addr32(&Bt[1][0])};

    auto stage = [&](int buf, int k0) {
        // A tile: 128 rows x 32 elems = 512 x 16B chunks (2 per thread)
#pragma unroll
        for (int t = 0; t < 2; ++t) {
            int c = tid + t * 256;
            int r = c >> 2, q = c & 3;
            async_copy_b128(as_base[buf] + r * (AS_S * 2) + q * 16,
                            A + (size_t)(m0 + r) * Kdim + k0 + q * 8);
        }
        // B tile: 64 rows x 32 elems = 256 chunks (1 per thread)
        {
            int r = tid >> 2, q = tid & 3;
            async_copy_b128(bt_base[buf] + r * (BT_S * 2) + q * 16,
                            Wt + (size_t)(n0 + r) * Kdim + k0 + q * 8);
        }
    };

    v8f acc[4];
#pragma unroll
    for (int j = 0; j < 4; ++j) acc[j] = (v8f){0.f,0.f,0.f,0.f,0.f,0.f,0.f,0.f};

    stage(0, 0);
    const int niter = Kdim / BK;
    for (int it = 0; it < niter; ++it) {
        const int cur = it & 1;
        const bool more = (it + 1) < niter;
        if (more) { stage(cur ^ 1, (it + 1) * BK); wait_async3(); }
        else      { wait_async0(); }
        __syncthreads();

        v16bf af;
#pragma unroll
        for (int i = 0; i < 8; ++i) {
            int kb = kbase(i, hi);
            put_pair(af, i, *(const unsigned*)&As[cur][(wave * 16 + m16) * AS_S + kb]);
        }
#pragma unroll
        for (int j = 0; j < 4; ++j) {
            v16bf bf;
#pragma unroll
            for (int i = 0; i < 8; ++i) {
                int kb = kbase(i, hi);
                put_pair(bf, i, *(const unsigned*)&Bt[cur][(j * 16 + m16) * BT_S + kb]);
            }
            acc[j] = wmma_bf16(af, bf, acc[j]);
        }
        __syncthreads();
    }

    // --- epilogue ---
#pragma unroll
    for (int j = 0; j < 4; ++j) {
        const int col = n0 + j * 16 + m16;
        const float bv = bias[col];
        const int row0 = m0 + wave * 16 + 8 * hi;     // 8 consecutive rows
        if (MODE == 2) {
            // QKV split; section/head uniform per wave (col groups 64-aligned)
            const int sec  = col >> 10;               // 0=q, 1=k, 2=v
            const int head = (col & 1023) >> 6;
            const int d    = col & 63;
            const int batch = row0 >> 11;             // S = 2048
            const int s0i   = row0 & 2047;
            const size_t bh = (size_t)(batch * NH + head);
            if (sec == 2) {
                __bf16 tmp[8];
#pragma unroll
                for (int r = 0; r < 8; ++r) tmp[r] = (__bf16)(acc[j][r] + bv);
                uint4 pk;
                __builtin_memcpy(&pk, tmp, 16);
                *(uint4*)((__bf16*)out2 + (bh * HD + d) * Ss + s0i) = pk;
            } else {
                __bf16* dst = ((sec == 0) ? (__bf16*)out0 : (__bf16*)out1) +
                              (bh * Ss + s0i) * HD + d;
#pragma unroll
                for (int r = 0; r < 8; ++r) dst[r * HD] = (__bf16)(acc[j][r] + bv);
            }
        } else {
#pragma unroll
            for (int r = 0; r < 8; ++r) {
                const int row = row0 + r;
                float v = acc[j][r] + bv;
                if (GELU)  v = gelu_tanh(v);
                if (RESID) v += resid[(size_t)row * Ndim + col];
                if (MODE == 1) ((__bf16*)out0)[(size_t)row * Ndim + col] = (__bf16)v;
                else           ((float*)out0)[(size_t)row * Ndim + col] = v;
            }
        }
    }
}

// ---------------------------------------------------------------------------
// Flash attention: block = 8 waves = 8 query tiles of one (batch, head).
// K tile from kh[b,h,s,d] (contiguous 4KB), V tile from vtr[b,h,d,s]
// (rows contiguous over keys) -- both staged via async B128, double-buffered.
// ---------------------------------------------------------------------------
constexpr int ATT_WAVES = 8;
constexpr int KS_S = 72;   // K LDS row stride elems: 144 B (16B multiple)
constexpr int VT_S = 40;   // Vt LDS row stride elems: 80 B (16B multiple)

__global__ __launch_bounds__(ATT_WAVES * 32)
void attn_kernel(const __bf16* __restrict__ qh, const __bf16* __restrict__ kh,
                 const __bf16* __restrict__ vtr, __bf16* __restrict__ attn_out) {
    __shared__ __align__(16) __bf16 Ks[2][32 * KS_S];    // [key][d]
    __shared__ __align__(16) __bf16 Vt[2][64 * VT_S];    // [d][key]
    __shared__ __align__(16) __bf16 Pbuf[ATT_WAVES][16 * 32];

    const int tid  = threadIdx.x;
    const int lane = tid & 31;
    const int wave = tid >> 5;
    const int bid  = blockIdx.x;
    const int qt    = (bid & 15) * ATT_WAVES + wave;  // 128 query tiles / 8
    const int head  = (bid >> 4) & 15;
    const int batch = bid >> 8;

    const int m  = lane & 15;
    const int hi = lane >> 4;
    const size_t bh = (size_t)(batch * NH + head);
    const unsigned ks_base[2] = {lds_addr32(&Ks[0][0]), lds_addr32(&Ks[1][0])};
    const unsigned vt_base[2] = {lds_addr32(&Vt[0][0]), lds_addr32(&Vt[1][0])};

    const __bf16* kp_all = kh + bh * Ss * HD;   // [s][d], contiguous
    const __bf16* vp_all = vtr + bh * HD * Ss;  // [d][s]

    __bf16* P = Pbuf[wave];

    auto stage_kv = [&](int buf, int kt) {
        // K tile: 32 keys x 64 d, whole tile contiguous: 256 chunks (1/thread)
        {
            int r = tid >> 3, q = tid & 7;
            async_copy_b128(ks_base[buf] + r * (KS_S * 2) + q * 16,
                            kp_all + (size_t)(kt * 32 + r) * HD + q * 8);
        }
        // V tile: 64 d-rows x 32 keys: 256 chunks (1/thread)
        {
            int r = tid >> 2, q = tid & 3;
            async_copy_b128(vt_base[buf] + r * (VT_S * 2) + q * 16,
                            vp_all + (size_t)r * Ss + kt * 32 + q * 8);
        }
    };

    // q fragments (d 0..31 and d 32..63), 32-bit global reads
    const __bf16* qp = qh + (bh * Ss + qt * 16 + m) * HD;
    v16bf aq0, aq1;
#pragma unroll
    for (int i = 0; i < 8; ++i) {
        int kb = kbase(i, hi);
        put_pair(aq0, i, *(const unsigned*)(qp + kb));
        put_pair(aq1, i, *(const unsigned*)(qp + 32 + kb));
    }

    float mrow[8], lrow[8];
#pragma unroll
    for (int r = 0; r < 8; ++r) { mrow[r] = -1e30f; lrow[r] = 0.f; }
    v8f o[4];
#pragma unroll
    for (int j = 0; j < 4; ++j) o[j] = (v8f){0.f,0.f,0.f,0.f,0.f,0.f,0.f,0.f};

    stage_kv(0, 0);
    constexpr int NT = Ss / 32;
    for (int kt = 0; kt < NT; ++kt) {
        const int cur = kt & 1;
        const bool more = (kt + 1) < NT;
        if (more) { stage_kv(cur ^ 1, kt + 1); wait_async2(); }
        else      { wait_async0(); }
        __syncthreads();

        // --- scores: s0 = keys 0..15, s1 = keys 16..31 (two K=32 WMMAs each) ---
        v8f s0 = (v8f){0.f,0.f,0.f,0.f,0.f,0.f,0.f,0.f};
        v8f s1 = s0;
#pragma unroll
        for (int kh2 = 0; kh2 < 2; ++kh2) {
            v16bf blo, bhi2;
#pragma unroll
            for (int i = 0; i < 8; ++i) {
                int kb = kbase(i, hi);
                put_pair(blo,  i, *(const unsigned*)&Ks[cur][(kh2 * 16 + m) * KS_S + kb]);
                put_pair(bhi2, i, *(const unsigned*)&Ks[cur][(kh2 * 16 + m) * KS_S + 32 + kb]);
            }
            if (kh2 == 0) { s0 = wmma_bf16(aq0, blo, s0); s0 = wmma_bf16(aq1, bhi2, s0); }
            else          { s1 = wmma_bf16(aq0, blo, s1); s1 = wmma_bf16(aq1, bhi2, s1); }
        }

        // --- online softmax update (DPP16 reductions, no LDS) ---
#pragma unroll
        for (int r = 0; r < 8; ++r) {
            float a0 = s0[r] * 0.125f;       // 1/sqrt(64)
            float a1 = s1[r] * 0.125f;
            float vmax = red16_max(fmaxf(a0, a1));
            float mnew  = fmaxf(mrow[r], vmax);
            float scale = __expf(mrow[r] - mnew);
            float p0 = __expf(a0 - mnew);
            float p1 = __expf(a1 - mnew);
            float psum = red16_sum(p0 + p1);
            lrow[r] = lrow[r] * scale + psum;
            mrow[r] = mnew;
            o[0][r] *= scale; o[1][r] *= scale; o[2][r] *= scale; o[3][r] *= scale;
            int prow = r + 8 * hi;
            P[prow * 32 + m]      = (__bf16)p0;
            P[prow * 32 + 16 + m] = (__bf16)p1;
        }

        // --- probs -> A fragment (wave-private LDS relayout) ---
        v16bf ap;
#pragma unroll
        for (int i = 0; i < 8; ++i) {
            int kb = kbase(i, hi);
            put_pair(ap, i, *(const unsigned*)&P[m * 32 + kb]);
        }

        // --- P @ V from Vt[d][key] (pairs contiguous over key) ---
#pragma unroll
        for (int j = 0; j < 4; ++j) {
            v16bf bv;
#pragma unroll
            for (int i = 0; i < 8; ++i) {
                int kb = kbase(i, hi);
                put_pair(bv, i, *(const unsigned*)&Vt[cur][(j * 16 + m) * VT_S + kb]);
            }
            o[j] = wmma_bf16(ap, bv, o[j]);
        }
        __syncthreads();
    }

    // --- normalize + store (heads recombined into [B*S][H] bf16) ---
#pragma unroll
    for (int j = 0; j < 4; ++j) {
#pragma unroll
        for (int r = 0; r < 8; ++r) {
            int row = batch * Ss + qt * 16 + r + 8 * hi;
            float v = o[j][r] / lrow[r];
            attn_out[(size_t)row * Hh + head * HD + j * 16 + m] = (__bf16)v;
        }
    }
}

// ---------------------------------------------------------------------------
// launch
// ---------------------------------------------------------------------------
extern "C" void kernel_launch(void* const* d_in, const int* in_sizes, int n_in,
                              void* d_out, int out_size, void* d_ws, size_t ws_size,
                              hipStream_t stream) {
    const float* x     = (const float*)d_in[0];
    const float* ln1_g = (const float*)d_in[1];
    const float* ln1_b = (const float*)d_in[2];
    const float* qkv_w = (const float*)d_in[3];
    const float* qkv_b = (const float*)d_in[4];
    const float* out_w = (const float*)d_in[5];
    const float* out_b = (const float*)d_in[6];
    const float* ln2_g = (const float*)d_in[7];
    const float* ln2_b = (const float*)d_in[8];
    const float* w1    = (const float*)d_in[9];
    const float* b1    = (const float*)d_in[10];
    const float* w2    = (const float*)d_in[11];
    const float* b2    = (const float*)d_in[12];
    float* out = (float*)d_out;

    // workspace carve-up (bytes)
    char* ws = (char*)d_ws;
    auto take = [&](size_t bytes) { char* p = ws; ws += (bytes + 255) & ~size_t(255); return p; };
    __bf16* h1_bf   = (__bf16*)take((size_t)Mrows * Hh * 2);
    __bf16* qh      = (__bf16*)take((size_t)Mrows * Hh * 2);   // [b,h,s,d]
    __bf16* khb     = (__bf16*)take((size_t)Mrows * Hh * 2);   // [b,h,s,d]
    __bf16* vtr     = (__bf16*)take((size_t)Mrows * Hh * 2);   // [b,h,d,s]
    __bf16* attn_bf = (__bf16*)take((size_t)Mrows * Hh * 2);
    float*  x1      = (float*) take((size_t)Mrows * Hh * 4);
    __bf16* h2_bf   = (__bf16*)take((size_t)Mrows * Hh * 2);
    __bf16* g_bf    = (__bf16*)take((size_t)Mrows * Ii * 2);
    __bf16* wqkv_t  = (__bf16*)take((size_t)Hh * 3 * Hh * 2);  // [3H][H]
    __bf16* wout_t  = (__bf16*)take((size_t)Hh * Hh * 2);      // [H][H]
    __bf16* w1_t    = (__bf16*)take((size_t)Hh * Ii * 2);      // [I][H]
    __bf16* w2_t    = (__bf16*)take((size_t)Ii * Hh * 2);      // [H][I]

    // 1) weight casts + transposes: W[K][N] -> Wt[N][K] bf16
    auto castT = [&](const float* src, __bf16* dst, int K, int N) {
        int n = K * N;
        cast_transpose_kernel<<<(n + 255) / 256, 256, 0, stream>>>(src, dst, K, N);
    };
    castT(qkv_w, wqkv_t, Hh, 3 * Hh);
    castT(out_w, wout_t, Hh, Hh);
    castT(w1, w1_t, Hh, Ii);
    castT(w2, w2_t, Ii, Hh);

    // 2) LN1
    ln_kernel<<<Mrows, 256, 0, stream>>>(x, ln1_g, ln1_b, h1_bf);

    // 3) QKV GEMM -> qh/kh [b,h,s,d] + vtr [b,h,d,s]
    gemm_wmma<2, false, false><<<dim3(3 * Hh / BN, Mrows / BM), 256, 0, stream>>>(
        h1_bf, wqkv_t, qkv_b, nullptr, qh, khb, vtr, Mrows, 3 * Hh, Hh);

    // 4) attention -> bf16 [M, H]
    attn_kernel<<<Bb * NH * (Ss / 16) / ATT_WAVES, ATT_WAVES * 32, 0, stream>>>(
        qh, khb, vtr, attn_bf);

    // 5) output projection + residual(x) -> f32 x1
    gemm_wmma<0, false, true><<<dim3(Hh / BN, Mrows / BM), 256, 0, stream>>>(
        attn_bf, wout_t, out_b, x, x1, nullptr, nullptr, Mrows, Hh, Hh);

    // 6) LN2
    ln_kernel<<<Mrows, 256, 0, stream>>>(x1, ln2_g, ln2_b, h2_bf);

    // 7) MLP up + GELU -> bf16 [M, I]
    gemm_wmma<1, true, false><<<dim3(Ii / BN, Mrows / BM), 256, 0, stream>>>(
        h2_bf, w1_t, b1, nullptr, g_bf, nullptr, nullptr, Mrows, Ii, Hh);

    // 8) MLP down + residual(x1) -> d_out f32
    gemm_wmma<0, false, true><<<dim3(Hh / BN, Mrows / BM), 256, 0, stream>>>(
        g_bf, w2_t, b2, x1, out, nullptr, nullptr, Mrows, Hh, Ii);
}
